// Encoder_14379550507205
// MI455X (gfx1250) — matrix operand order repaired and checked by reading, not compiled, not run on hardware
//
#include <hip/hip_runtime.h>
#include <hip/hip_bf16.h>
#include <stdint.h>
#include <stddef.h>

// ---------------- problem constants (match reference) ----------------
constexpr int B = 4, S = 2048, D = 1024, H = 16, DH = 64;
constexpr float NEGF  = -1e9f;
constexpr float SCALE = 0.125f;   // 1/sqrt(DH) = 1/8

// ---------------- WMMA types (CDNA5 gfx1250, wave32) ----------------
typedef __attribute__((ext_vector_type(16))) _Float16 v16h;
typedef __attribute__((ext_vector_type(8)))  float    v8f;

union Frag {            // one 16x32 A-fragment or 32x16 B-fragment (16 f16 / lane)
  v16h  v;
  uint4 q[2];
};
union H8 {              // 8 f16 packed for one uint4 LDS store
  _Float16 h[8];
  uint4    q;
};

__device__ inline v8f wmma_f16(const Frag& a, const Frag& b, v8f c) {
  // D = A(16x32) * B(32x16) + C, f32 accumulate  -> v_wmma_f32_16x16x32_f16
  return __builtin_amdgcn_wmma_f32_16x16x32_f16(
      /*neg_a=*/false, a.v, /*neg_b=*/false, b.v,
      /*c_mod=*/(short)0, c, /*reuse_a=*/false, /*reuse_b=*/false);
}

__device__ inline uint4 cvt8(const float4 f0, const float4 f1) {
  H8 p;
  p.h[0] = (_Float16)f0.x; p.h[1] = (_Float16)f0.y;
  p.h[2] = (_Float16)f0.z; p.h[3] = (_Float16)f0.w;
  p.h[4] = (_Float16)f1.x; p.h[5] = (_Float16)f1.y;
  p.h[6] = (_Float16)f1.z; p.h[7] = (_Float16)f1.w;
  return p.q;
}

// ---------------- CDNA5 Tensor Data Mover (TDM) support ----------------
#ifndef __has_builtin
#define __has_builtin(x) 0
#endif
#if __has_builtin(__builtin_amdgcn_tensor_load_to_lds) && \
    __has_builtin(__builtin_amdgcn_s_wait_tensorcnt)
#define USE_TDM 1
#else
#define USE_TDM 0
#endif
#if __has_include(<hip/amd_detail/amd_gfx1250_TDM.h>)
#define TDM_SIX_ARGS 1   // therock-10.0 headers -> 6-arg builtin
#else
#define TDM_SIX_ARGS 0   // ROCm 7.2 -> 5-arg builtin
#endif

// Low 32 bits of a flat __shared__ pointer == wave-relative LDS byte address
// (flat->LDS aperture mapping truncates to addr[31:0], ISA 10.2).
__device__ inline unsigned lds_byte_addr(const void* p) {
  return (unsigned)(size_t)p;
}

#if USE_TDM
typedef __attribute__((ext_vector_type(4))) unsigned int u32x4;
typedef __attribute__((ext_vector_type(8))) int          i32x8;
typedef __attribute__((ext_vector_type(4))) int          i32x4;

// DMA one 64x64 f16 tile (row stride DH in memory) into LDS whose rows are
// padded to 72 halfs: 32 dwords of data + 4 dwords pad per row, expressed via
// D# pad_enable (pad_interval code 4 == 32 dwords, pad_amount code 3 == 4 dw).
__device__ inline void tdm_load_tile64(const _Float16* gsrc, unsigned ldsByteAddr) {
  const unsigned long long ga = (unsigned long long)(size_t)gsrc;
  u32x4 g0;
  g0.x = 1u;                                            // count=1 (valid user D#)
  g0.y = ldsByteAddr;                                   // lds_addr
  g0.z = (unsigned)ga;                                  // global_addr[31:0]
  g0.w = (unsigned)((ga >> 32) & 0x01FFFFFFull)         // global_addr[56:32]
       | (2u << 30);                                    // type=2 ("image")
  i32x8 g1;
  g1[0] = (1 << 16)      // data_size = 2 bytes
        | (1 << 20)      // pad_enable
        | (4 << 22)      // pad_interval: 32 dwords between pads
        | (3 << 25);     // pad_amount : 4 dwords of pad
  g1[1] = (DH << 16);          // tensor_dim0[15:0] = 64 (bits 63:48)
  g1[2] = (int)(2048u << 16);  // tensor_dim1[15:0] = S  (bits 111:96->? packed LE)
  g1[3] = (DH << 16);          // tile_dim0 = 64 (bits 127:112)
  g1[4] = 64;                  // tile_dim1 = 64 rows; tile_dim2 = 0
  g1[5] = DH;                  // tensor_dim0_stride = 64 elements
  g1[6] = 0;
  g1[7] = 0;
  const i32x4 z4 = {0, 0, 0, 0};
#if TDM_SIX_ARGS
  const i32x8 z8 = {0, 0, 0, 0, 0, 0, 0, 0};
  __builtin_amdgcn_tensor_load_to_lds(g0, g1, z4, z4, z8, 0);
#else
  __builtin_amdgcn_tensor_load_to_lds(g0, g1, z4, z4, 0);
#endif
}
#endif  // USE_TDM

// =====================================================================
// Kernel 1: fused QKV projection.  y = x @ W^T + b, stored f16 [B,H,S,DH]
//   grid = (B*S/128, 3*H), block = 256 (8 waves)
// =====================================================================
__global__ __launch_bounds__(256)
void qkv_proj_kernel(const float* __restrict__ seq,
                     const float* __restrict__ Wq, const float* __restrict__ bq,
                     const float* __restrict__ Wk, const float* __restrict__ bk,
                     const float* __restrict__ Wv, const float* __restrict__ bv,
                     _Float16* __restrict__ Qws,
                     _Float16* __restrict__ Kws,
                     _Float16* __restrict__ Vws) {
  __shared__ __align__(16) _Float16 lA[128][40];  // 128x32 tokens tile (+pad)
  __shared__ __align__(16) _Float16 lB[64][40];   // 64x32 weight tile  (+pad)

  const int t    = threadIdx.x;
  const int mat  = blockIdx.y / H;   // 0=Q 1=K 2=V
  const int head = blockIdx.y % H;

  const float* Wm; const float* bm; _Float16* om;
  if (mat == 0)      { Wm = Wq; bm = bq; om = Qws; }
  else if (mat == 1) { Wm = Wk; bm = bk; om = Kws; }
  else               { Wm = Wv; bm = bv; om = Vws; }

  const int m_base = blockIdx.x * 128;
  const int bidx   = m_base / S;
  const int s_base = m_base % S;

  const int lane = t & 31;
  const int wv   = t >> 5;
  const int r    = lane & 15;
  const int hi   = lane >> 4;
  const int wm   = (wv & 3) * 32;
  const int wn   = (wv >> 2) * 32;

  v8f c00 = {}, c01 = {}, c10 = {}, c11 = {};

  const int arow = t >> 1, acb = (t & 1) * 16;
  const float* aSrc = seq + (size_t)(m_base + arow) * D + acb;
  const int brow = (t & 127) >> 1, bcb = ((t & 127) & 1) * 16;
  const float* bSrc = Wm + (size_t)(head * DH + brow) * D + bcb;

  for (int kk = 0; kk < D; kk += 32) {
    { // stage A: 128x32 f32 -> f16 in LDS
      const float4 f0 = *(const float4*)(aSrc + kk + 0);
      const float4 f1 = *(const float4*)(aSrc + kk + 4);
      const float4 f2 = *(const float4*)(aSrc + kk + 8);
      const float4 f3 = *(const float4*)(aSrc + kk + 12);
      *(uint4*)&lA[arow][acb]     = cvt8(f0, f1);
      *(uint4*)&lA[arow][acb + 8] = cvt8(f2, f3);
      if (kk + 32 < D) __builtin_prefetch(aSrc + kk + 32, 0, 0); // global_prefetch_b8
    }
    if (t < 128) { // stage B: 64x32 rows of W
      const float4 f0 = *(const float4*)(bSrc + kk + 0);
      const float4 f1 = *(const float4*)(bSrc + kk + 4);
      const float4 f2 = *(const float4*)(bSrc + kk + 8);
      const float4 f3 = *(const float4*)(bSrc + kk + 12);
      *(uint4*)&lB[brow][bcb]     = cvt8(f0, f1);
      *(uint4*)&lB[brow][bcb + 8] = cvt8(f2, f3);
      if (kk + 32 < D) __builtin_prefetch(bSrc + kk + 32, 0, 0);
    }
    __syncthreads();

    Frag a0, a1, b0, b1;
    a0.q[0] = *(const uint4*)&lA[wm + r][hi * 8];
    a0.q[1] = *(const uint4*)&lA[wm + r][16 + hi * 8];
    a1.q[0] = *(const uint4*)&lA[wm + 16 + r][hi * 8];
    a1.q[1] = *(const uint4*)&lA[wm + 16 + r][16 + hi * 8];
    b0.q[0] = *(const uint4*)&lB[wn + r][hi * 8];
    b0.q[1] = *(const uint4*)&lB[wn + r][16 + hi * 8];
    b1.q[0] = *(const uint4*)&lB[wn + 16 + r][hi * 8];
    b1.q[1] = *(const uint4*)&lB[wn + 16 + r][16 + hi * 8];

    c00 = wmma_f16(a0, b0, c00);
    c01 = wmma_f16(a0, b1, c01);
    c10 = wmma_f16(a1, b0, c10);
    c11 = wmma_f16(a1, b1, c11);
    __syncthreads();
  }

  const float bias0 = bm[head * DH + wn + r];
  const float bias1 = bm[head * DH + wn + 16 + r];
  _Float16* outBase = om + (size_t)(bidx * H + head) * S * DH;
#pragma unroll
  for (int j = 0; j < 8; ++j) {
    const int row0 = s_base + wm + hi * 8 + j;
    const int row1 = row0 + 16;
    outBase[(size_t)row0 * DH + wn + r]      = (_Float16)(c00[j] + bias0);
    outBase[(size_t)row0 * DH + wn + 16 + r] = (_Float16)(c01[j] + bias1);
    outBase[(size_t)row1 * DH + wn + r]      = (_Float16)(c10[j] + bias0);
    outBase[(size_t)row1 * DH + wn + 16 + r] = (_Float16)(c11[j] + bias1);
  }
}

// =====================================================================
// Kernel 2: flash attention per (b,h).
//   grid = (S/64, B*H), block = 128 (4 waves)
//   Q + K tiles staged by the Tensor Data Mover (double-buffered K),
//   V transposed into LDS by the wave (TDM cannot transpose).
// =====================================================================
__global__ __launch_bounds__(128)
void attn_kernel(const _Float16* __restrict__ Qws,
                 const _Float16* __restrict__ Kws,
                 const _Float16* __restrict__ Vws,
                 const unsigned char* __restrict__ mask,  // jnp bool -> 1 byte
                 float* __restrict__ out) {
  __shared__ __align__(16) _Float16 sQ [64][72];        // [qrow][dh]
  __shared__ __align__(16) _Float16 sK [2][64][72];     // [buf][key][dh]
  __shared__ __align__(16) _Float16 sVT[64][72];        // [dh][key] (transposed)
  __shared__ __align__(16) _Float16 sP [4][16][72];     // per-wave P tile
  __shared__ float qmNeg[64];
  __shared__ float kmNeg[64];

  const int t  = threadIdx.x;
  const int bh = blockIdx.y;
  const int b  = bh / H, h = bh % H;
  const int qbase = blockIdx.x * 64;
  const int NT = S / 64;

  const _Float16* Qg = Qws + (size_t)bh * S * DH;
  const _Float16* Kg = Kws + (size_t)bh * S * DH;
  const _Float16* Vg = Vws + (size_t)bh * S * DH;

#if USE_TDM
  if (t == 0) {  // wave 0 drives the TDM (EXEC ignored by tensor ops)
    tdm_load_tile64(Qg + (size_t)qbase * DH, lds_byte_addr(&sQ[0][0]));
    tdm_load_tile64(Kg,                      lds_byte_addr(&sK[0][0][0]));
    __builtin_amdgcn_s_wait_tensorcnt(0);    // s_wait_tensorcnt 0
  }
#else
  { // manual Q stage (64x64 f16)
    const int row = t >> 1, cb = (t & 1) * 32;
    const uint4* src = (const uint4*)(Qg + (size_t)(qbase + row) * DH + cb);
    uint4 v0 = src[0], v1 = src[1], v2 = src[2], v3 = src[3];
    *(uint4*)&sQ[row][cb + 0]  = v0;
    *(uint4*)&sQ[row][cb + 8]  = v1;
    *(uint4*)&sQ[row][cb + 16] = v2;
    *(uint4*)&sQ[row][cb + 24] = v3;
  }
#endif
  if (t < 64) qmNeg[t] = mask[(size_t)b * S + qbase + t] ? NEGF : 0.0f;
  __syncthreads();

  const int lane = t & 31, wv = t >> 5;
  const int r = lane & 15, hi = lane >> 4;
  const int qr0 = wv * 16;

  Frag aq0, aq1;  // Q A-fragments, d = [0,32) and [32,64)
  aq0.q[0] = *(const uint4*)&sQ[qr0 + r][0  + hi * 8];
  aq0.q[1] = *(const uint4*)&sQ[qr0 + r][16 + hi * 8];
  aq1.q[0] = *(const uint4*)&sQ[qr0 + r][32 + hi * 8];
  aq1.q[1] = *(const uint4*)&sQ[qr0 + r][48 + hi * 8];

  float qn[8];
#pragma unroll
  for (int j = 0; j < 8; ++j) qn[j] = qmNeg[qr0 + hi * 8 + j];

  v8f o0 = {}, o1 = {}, o2 = {}, o3 = {};
  float mrow[8], lrow[8];
#pragma unroll
  for (int j = 0; j < 8; ++j) { mrow[j] = -1e30f; lrow[j] = 0.0f; }

  for (int kt = 0; kt < NT; ++kt) {
    const int ktb = kt * 64;
    const int cur = kt & 1;

#if USE_TDM
    // prefetch next K tile into the other buffer while we stage V/compute
    if (t == 0 && kt + 1 < NT)
      tdm_load_tile64(Kg + (size_t)(ktb + 64) * DH,
                      lds_byte_addr(&sK[(kt + 1) & 1][0][0]));
#else
    { // manual K stage into sK[cur]
      const int row = t >> 1, cb = (t & 1) * 32;
      const uint4* ks = (const uint4*)(Kg + (size_t)(ktb + row) * DH + cb);
      uint4 k0 = ks[0], k1 = ks[1], k2 = ks[2], k3 = ks[3];
      *(uint4*)&sK[cur][row][cb + 0]  = k0;
      *(uint4*)&sK[cur][row][cb + 8]  = k1;
      *(uint4*)&sK[cur][row][cb + 16] = k2;
      *(uint4*)&sK[cur][row][cb + 24] = k3;
    }
#endif
    { // stage V transposed (all 128 threads, 32 f16 each)
      const int row = t >> 1, cb = (t & 1) * 32;
      const uint4* vs = (const uint4*)(Vg + (size_t)(ktb + row) * DH + cb);
      uint4 w[4] = { vs[0], vs[1], vs[2], vs[3] };
#pragma unroll
      for (int i = 0; i < 4; ++i) {
        unsigned int xs[4] = { w[i].x, w[i].y, w[i].z, w[i].w };
#pragma unroll
        for (int e = 0; e < 4; ++e) {
          const int dh0 = cb + i * 8 + e * 2;
          H8 tmp; tmp.q = uint4{xs[e], 0u, 0u, 0u};
          sVT[dh0][row]     = tmp.h[0];
          sVT[dh0 + 1][row] = tmp.h[1];
        }
      }
    }
    if (t < 64) kmNeg[t] = mask[(size_t)b * S + ktb + t] ? 0.0f : NEGF;

#if USE_TDM
    if (t == 0) {  // TDM completes in-order: <=1 outstanding => current tile done
      if (kt + 1 < NT) __builtin_amdgcn_s_wait_tensorcnt(1);
      else             __builtin_amdgcn_s_wait_tensorcnt(0);
    }
#endif
    __syncthreads();

    // ---- scores: Sc = Q @ K^T (wave: 16 rows x 64 keys) ----
    v8f sc[4];
#pragma unroll
    for (int c = 0; c < 4; ++c) {
      Frag bk0, bk1;   // B frag: lane r = key col, d contiguous
      bk0.q[0] = *(const uint4*)&sK[cur][c * 16 + r][0  + hi * 8];
      bk0.q[1] = *(const uint4*)&sK[cur][c * 16 + r][16 + hi * 8];
      bk1.q[0] = *(const uint4*)&sK[cur][c * 16 + r][32 + hi * 8];
      bk1.q[1] = *(const uint4*)&sK[cur][c * 16 + r][48 + hi * 8];
      v8f acc = {};
      acc = wmma_f16(aq0, bk0, acc);
      acc = wmma_f16(aq1, bk1, acc);
      sc[c] = acc;
    }

    // ---- scale + mask + row-max (16-lane xor reduction) ----
    float x[4][8], km[4], rmax[8];
#pragma unroll
    for (int c = 0; c < 4; ++c) km[c] = kmNeg[c * 16 + r];
#pragma unroll
    for (int j = 0; j < 8; ++j) rmax[j] = -1e30f;
#pragma unroll
    for (int c = 0; c < 4; ++c)
#pragma unroll
      for (int j = 0; j < 8; ++j) {
        float v = sc[c][j] * SCALE;
        v = (km[c] != 0.0f) ? NEGF : v;
        v = (qn[j] != 0.0f) ? NEGF : v;
        x[c][j] = v;
        rmax[j] = fmaxf(rmax[j], v);
      }
#pragma unroll
    for (int j = 0; j < 8; ++j)
#pragma unroll
      for (int d2 = 1; d2 < 16; d2 <<= 1)
        rmax[j] = fmaxf(rmax[j], __shfl_xor(rmax[j], d2, 32));

    // ---- online softmax update ----
    float alpha[8], rs[8];
#pragma unroll
    for (int j = 0; j < 8; ++j) {
      const float mn = fmaxf(mrow[j], rmax[j]);
      alpha[j] = __expf(mrow[j] - mn);
      mrow[j]  = mn;
      rs[j]    = 0.0f;
    }
#pragma unroll
    for (int c = 0; c < 4; ++c)
#pragma unroll
      for (int j = 0; j < 8; ++j) {
        const float p = __expf(x[c][j] - mrow[j]);
        x[c][j] = p;
        rs[j] += p;
      }
#pragma unroll
    for (int j = 0; j < 8; ++j) {
#pragma unroll
      for (int d2 = 1; d2 < 16; d2 <<= 1)
        rs[j] += __shfl_xor(rs[j], d2, 32);
      lrow[j] = lrow[j] * alpha[j] + rs[j];
    }
#pragma unroll
    for (int j = 0; j < 8; ++j) {
      o0[j] *= alpha[j]; o1[j] *= alpha[j];
      o2[j] *= alpha[j]; o3[j] *= alpha[j];
    }

    // ---- restage P as f16 into per-wave LDS (C-layout -> A-layout) ----
#pragma unroll
    for (int c = 0; c < 4; ++c)
#pragma unroll
      for (int j = 0; j < 8; ++j)
        sP[wv][hi * 8 + j][c * 16 + r] = (_Float16)x[c][j];

    // ---- O += P @ V  (k = 64 keys, two 32-deep WMMA steps) ----
#pragma unroll
    for (int kd = 0; kd < 2; ++kd) {
      Frag ap;
      ap.q[0] = *(const uint4*)&sP[wv][r][kd * 32 + hi * 8];
      ap.q[1] = *(const uint4*)&sP[wv][r][kd * 32 + 16 + hi * 8];
      Frag bv0, bv1, bv2, bv3;  // B frag from V^T: lane r = dh col, key contiguous
      bv0.q[0] = *(const uint4*)&sVT[0  + r][kd * 32 + hi * 8];
      bv0.q[1] = *(const uint4*)&sVT[0  + r][kd * 32 + 16 + hi * 8];
      bv1.q[0] = *(const uint4*)&sVT[16 + r][kd * 32 + hi * 8];
      bv1.q[1] = *(const uint4*)&sVT[16 + r][kd * 32 + 16 + hi * 8];
      bv2.q[0] = *(const uint4*)&sVT[32 + r][kd * 32 + hi * 8];
      bv2.q[1] = *(const uint4*)&sVT[32 + r][kd * 32 + 16 + hi * 8];
      bv3.q[0] = *(const uint4*)&sVT[48 + r][kd * 32 + hi * 8];
      bv3.q[1] = *(const uint4*)&sVT[48 + r][kd * 32 + 16 + hi * 8];
      o0 = wmma_f16(ap, bv0, o0);
      o1 = wmma_f16(ap, bv1, o1);
      o2 = wmma_f16(ap, bv2, o2);
      o3 = wmma_f16(ap, bv3, o3);
    }
    __syncthreads();   // protect sVT/kmNeg (and K buffer rotation)
  }

  // ---- normalize and write out [B,S,D] fp32 ----
  float* ob = out + (size_t)b * S * D + h * DH;
#pragma unroll
  for (int j = 0; j < 8; ++j) {
    const float inv = 1.0f / lrow[j];
    const size_t row = (size_t)(qbase + qr0 + hi * 8 + j);
    ob[row * D + 0  + r] = o0[j] * inv;
    ob[row * D + 16 + r] = o1[j] * inv;
    ob[row * D + 32 + r] = o2[j] * inv;
    ob[row * D + 48 + r] = o3[j] * inv;
  }
}

// =====================================================================
// Host launch. Inputs (setup_inputs order):
//   0 seq [B,S,D] f32 | 1 mask [B,S] bool(1B) | 2 Wq | 3 bq | 4 Wk | 5 bk
//   6 Wv | 7 bv.  Workspace: Q,K,V f16 tensors [B,H,S,DH] = 48 MB total.
// =====================================================================
extern "C" void kernel_launch(void* const* d_in, const int* in_sizes, int n_in,
                              void* d_out, int out_size, void* d_ws, size_t ws_size,
                              hipStream_t stream) {
  const float*         seq  = (const float*)d_in[0];
  const unsigned char* mask = (const unsigned char*)d_in[1];
  const float* Wq = (const float*)d_in[2];
  const float* bq = (const float*)d_in[3];
  const float* Wk = (const float*)d_in[4];
  const float* bk = (const float*)d_in[5];
  const float* Wv = (const float*)d_in[6];
  const float* bv = (const float*)d_in[7];
  float* out = (float*)d_out;

  const size_t perTensor = (size_t)B * H * S * DH;   // 8,388,608 f16 elems
  _Float16* Qws = (_Float16*)d_ws;
  _Float16* Kws = Qws + perTensor;
  _Float16* Vws = Kws + perTensor;

  dim3 g1((B * S) / 128, 3 * H);   // (64, 48)
  qkv_proj_kernel<<<g1, dim3(256), 0, stream>>>(seq, Wq, bq, Wk, bk, Wv, bv,
                                                Qws, Kws, Vws);

  dim3 g2(S / 64, B * H);          // (32, 64)
  attn_kernel<<<g2, dim3(128), 0, stream>>>(Qws, Kws, Vws, mask, out);
}
